// scGPT_50397146251455
// MI455X (gfx1250) — compile-verified, matrix operand order checked by reading, not run on hardware
//
#include <hip/hip_runtime.h>
#include <hip/hip_bf16.h>

// ---------------- model constants ----------------
constexpr int Lc   = 6;
constexpr int Dc   = 512;
constexpr int Hc   = 8;
constexpr int FFc  = 2048;
constexpr int Vc   = 20000;
constexpr int Vpad = 20096;      // next multiple of 128
constexpr int Bc   = 8;
constexpr int Sc   = 512;
constexpr int DHc  = 64;         // D / H
constexpr int Mrows = Bc * Sc;   // 4096

// ---------------- helpers ----------------
typedef __attribute__((ext_vector_type(16))) __bf16 v16bf;
typedef __attribute__((ext_vector_type(8)))  float  v8f;

__device__ __host__ inline unsigned short f32_to_bf16u(float f) {
  union { float f; unsigned u; } c; c.f = f;
  unsigned u = c.u;
  return (unsigned short)((u + 0x7FFFu + ((u >> 16) & 1u)) >> 16);
}

// ---------------- fp32 -> bf16 with transpose: out[n*K + k] = bf16(in[k*N + n]) ----------
// Makes every GEMM B operand k-contiguous so tile staging is pure b128 copies.
__global__ __launch_bounds__(256) void f2bf_t_kernel(const float* __restrict__ in,
                                                     unsigned short* __restrict__ out,
                                                     int K, int N) {
  const int k = blockIdx.x * 256 + threadIdx.x;
  const int n = blockIdx.y;
  if (k < K) out[(size_t)n * K + k] = f32_to_bf16u(in[(size_t)k * N + n]);
}

// ---------------- embedding + expr-MLP + pos ----------------
__global__ __launch_bounds__(128) void embed_kernel(
    const int* __restrict__ gene, const float* __restrict__ ev,
    const int* __restrict__ cond, const int* __restrict__ bat,
    const int* __restrict__ mod,
    const float* __restrict__ gene_emb, const float* __restrict__ cond_emb,
    const float* __restrict__ bat_emb,  const float* __restrict__ mod_emb,
    const float* __restrict__ w1, const float* __restrict__ b1,
    const float* __restrict__ w2, const float* __restrict__ b2,
    const float* __restrict__ pos,
    float* __restrict__ xf, unsigned short* __restrict__ xbf) {
  const int tok = blockIdx.x;          // 0 .. B*S-1
  const int s = tok & (Sc - 1);
  __shared__ float hbuf[Dc / 2];       // 256 hidden units
  const float e = ev[tok];
  for (int j = threadIdx.x; j < Dc / 2; j += blockDim.x)
    hbuf[j] = fmaxf(e * w1[j] + b1[j], 0.f);
  __syncthreads();
  const int g = gene[tok], c = cond[tok], bb = bat[tok], mm = mod[tok];
  for (int d = threadIdx.x; d < Dc; d += blockDim.x) {
    float acc = b2[d];
    for (int j = 0; j < Dc / 2; ++j) acc = fmaf(hbuf[j], w2[j * Dc + d], acc);
    float val = gene_emb[(size_t)g * Dc + d] + cond_emb[(size_t)c * Dc + d]
              + bat_emb[(size_t)bb * Dc + d] + mod_emb[(size_t)mm * Dc + d]
              + acc + pos[(size_t)s * Dc + d];
    xf[(size_t)tok * Dc + d]  = val;
    xbf[(size_t)tok * Dc + d] = f32_to_bf16u(val);
  }
}

// ---------------- WMMA bf16 GEMM: C[M,N] = act(A[M,K] @ Bt^T + bias) ----------------
// A:  [M x K] bf16 row-major.  Bt: [Npad x K] bf16 row-major (pre-transposed weights).
// block = 256 threads (8 waves), block tile 64x128, K-step 32.
// wave w: rows wm=(w>>1)*16, cols wn=(w&1)*64; 1 A-frag x 4 B-frags = 4 WMMA/iter.
// Both tiles staged k-contiguous via GLOBAL_LOAD_ASYNC_TO_LDS_B128 (ASYNCcnt),
// every CDNA5 bf16 fragment is then exactly two ds_load_b128 per lane:
//   elements 0..7  <- k = lh*8 .. lh*8+7,  elements 8..15 <- k = lh*8+16 .. +23
__global__ __launch_bounds__(256) void gemm_bf16_kernel(
    const unsigned short* __restrict__ A, const unsigned short* __restrict__ Bt,
    const float* __restrict__ bias,
    float* __restrict__ Cf, unsigned short* __restrict__ Cbf,
    int M, int N, int K, int relu_act) {
  constexpr int AS = 40;                    // padded k-stride (bank dodge, 16B aligned)
  __shared__ unsigned short As[64 * AS];    // 64 rows x 32 k
  __shared__ unsigned short Bs[128 * AS];   // 128 cols x 32 k
  const int tid = threadIdx.x;
  const int wv = tid >> 5, lane = tid & 31;
  const int bm = blockIdx.y << 6, bn = blockIdx.x << 7;
  const int wm = (wv >> 1) << 4;            // 0,16,32,48
  const int wn = (wv & 1) << 6;             // 0,64
  const int lh = lane >> 4;
  const int rr = lane & 15;

  // staging assignments (b128 per async op)
  const int ia = tid << 3;                  // A tile: 2048 elems, 8/thread
  const int ar = ia >> 5, ac = ia & 31;
  const int bcol = tid >> 1;                // B tile: 128 cols x 32 k, 16/thread
  const int bko  = (tid & 1) << 4;          // 0 or 16

  const unsigned lds_a = (unsigned)(size_t)(void*)(As + ar * AS + ac);
  const unsigned lds_b = (unsigned)(size_t)(void*)(Bs + bcol * AS + bko);

  const v8f vzero = {0.f,0.f,0.f,0.f,0.f,0.f,0.f,0.f};
  v8f accs[4] = {vzero, vzero, vzero, vzero};

  union FragU { uint4 q[2]; v16bf v; };

  for (int k0 = 0; k0 < K; k0 += 32) {
    // async stage A (1 x b128) and B (2 x b128) directly into LDS
    const unsigned short* ga = A  + (size_t)(bm + ar)   * K + (k0 + ac);
    const unsigned short* gb = Bt + (size_t)(bn + bcol) * K + (k0 + bko);
    asm volatile("global_load_async_to_lds_b128 %0, %1, off"
                 :: "v"(lds_a), "v"(ga) : "memory");
    asm volatile("global_load_async_to_lds_b128 %0, %1, off"
                 :: "v"(lds_b), "v"(gb) : "memory");
    asm volatile("global_load_async_to_lds_b128 %0, %1, off offset:16"
                 :: "v"(lds_b), "v"(gb) : "memory");
    // prefetch next K-slab while the async DMA runs
    if (k0 + 32 < K) {
      __builtin_prefetch(ga + 32, 0, 1);
      __builtin_prefetch(gb + 32, 0, 1);
    }
    asm volatile("s_wait_asynccnt 0x0" ::: "memory");
    __syncthreads();
    // fragments: 2x ds_load_b128 each
    FragU fa;
    const int abase = (wm + rr) * AS + (lh << 3);
    fa.q[0] = *(const uint4*)(As + abase);
    fa.q[1] = *(const uint4*)(As + abase + 16);
#pragma unroll
    for (int t = 0; t < 4; ++t) {
      FragU fb;
      const int bbase = (wn + t * 16 + rr) * AS + (lh << 3);
      fb.q[0] = *(const uint4*)(Bs + bbase);
      fb.q[1] = *(const uint4*)(Bs + bbase + 16);
      accs[t] = __builtin_amdgcn_wmma_f32_16x16x32_bf16(
          false, fa.v, false, fb.v, (short)0, accs[t], false, false);
    }
    __syncthreads();
  }
  // epilogue: C/D layout: VGPR j -> row = j + lh*8, col = lane&15
#pragma unroll
  for (int t = 0; t < 4; ++t) {
    const int nc = bn + wn + t * 16 + rr;
    if (nc < N) {
      const float bsv = bias[nc];
#pragma unroll
      for (int j = 0; j < 8; ++j) {
        const int mr = bm + wm + j + (lh << 3);
        float vo = accs[t][j] + bsv;
        if (relu_act) vo = fmaxf(vo, 0.f);
        if (Cf)  Cf[(size_t)mr * N + nc] = vo;
        if (Cbf) Cbf[(size_t)mr * N + nc] = f32_to_bf16u(vo);
      }
    }
  }
  (void)M;
}

// ---------------- attention: one wave owns one (b,n) row across all heads ----------------
// scores[n,m] = (q[n].k[m]) / (sum_d k[n,d] + 1e-6), masked softmax, o = attn @ v.
// am row sums to 1 => thr = am.mean() = 1/S exactly; next-layer mask fused here.
__global__ __launch_bounds__(256) void attn_kernel(
    const float* __restrict__ q, const float* __restrict__ k,
    const float* __restrict__ v,
    unsigned char* __restrict__ mask, int use_mask,
    unsigned short* __restrict__ o_bf) {
  __shared__ float arow[8][Sc];
  __shared__ float amrow[8][Sc];
  const int tid = threadIdx.x, wv = tid >> 5, lane = tid & 31;
  const int row = blockIdx.x * 8 + wv;     // b*S + n
  const int n = row & (Sc - 1);
  const int b = row >> 9;                  // S == 512
  unsigned char* mrow = mask + (size_t)row * Sc;

  for (int m = lane; m < Sc; m += 32) amrow[wv][m] = 0.f;

  for (int h = 0; h < Hc; ++h) {
    const float* qrow = q + (size_t)row * Dc + h * DHc;
    float qreg[DHc];
#pragma unroll
    for (int d = 0; d < DHc; d += 4) {
      const float4 q4 = *(const float4*)(qrow + d);
      qreg[d] = q4.x; qreg[d + 1] = q4.y; qreg[d + 2] = q4.z; qreg[d + 3] = q4.w;
    }
    // row temperature: sum of k[b,h,n,:]
    const float* krn = k + (size_t)row * Dc + h * DHc;
    float ks = krn[lane] + krn[lane + 32];
    for (int off = 16; off; off >>= 1) ks += __shfl_xor(ks, off);
    const float inv = 1.0f / (ks + 1e-6f);

    float mx = -1e30f;
    for (int m = lane; m <= n; m += 32) {
      const bool allowed = use_mask ? (mrow[m] != 0) : true;
      float sres;
      if (allowed) {
        const float* krow = k + (size_t)(b * Sc + m) * Dc + h * DHc;
        float acc = 0.f;
#pragma unroll 8
        for (int d = 0; d < DHc; ++d) acc = fmaf(qreg[d], krow[d], acc);
        sres = acc * inv;
        mx = fmaxf(mx, sres);
      } else {
        sres = -INFINITY;
      }
      arow[wv][m] = sres;
    }
    for (int off = 16; off; off >>= 1) mx = fmaxf(mx, __shfl_xor(mx, off));

    float ssum = 0.f;
    for (int m = lane; m <= n; m += 32) {
      float e = __expf(arow[wv][m] - mx);
      arow[wv][m] = e;
      ssum += e;
    }
    for (int off = 16; off; off >>= 1) ssum += __shfl_xor(ssum, off);
    const float isum = 1.0f / ssum;
    for (int m = lane; m <= n; m += 32) {
      float p = arow[wv][m] * isum;
      arow[wv][m] = p;
      amrow[wv][m] += p * (1.0f / Hc);
    }
    // o[n, h*DH + d]; lane owns d = lane, lane+32
    float a0 = 0.f, a1 = 0.f;
    for (int m = 0; m <= n; ++m) {
      const float p = arow[wv][m];
      const float* vrow = v + (size_t)(b * Sc + m) * Dc + h * DHc;
      a0 = fmaf(p, vrow[lane], a0);
      a1 = fmaf(p, vrow[lane + 32], a1);
    }
    unsigned short* orow = o_bf + (size_t)row * Dc + h * DHc;
    orow[lane]      = f32_to_bf16u(a0);
    orow[lane + 32] = f32_to_bf16u(a1);
  }
  // next-layer mask: max(uncover*causal, eye); thr == 1/S exactly
  const float thr = 1.0f / (float)Sc;
  for (int m = lane; m < Sc; m += 32) {
    const bool allowed = (m == n) || (m <= n && amrow[wv][m] > thr);
    mrow[m] = allowed ? (unsigned char)1 : (unsigned char)0;
  }
}

// ---------------- residual add + LayerNorm (one wave per row) ----------------
__global__ __launch_bounds__(256) void add_ln_kernel(
    const float* __restrict__ x, const float* __restrict__ r,
    const float* __restrict__ g, const float* __restrict__ bta,
    float* __restrict__ xf, unsigned short* __restrict__ xbf) {
  const int tid = threadIdx.x, wv = tid >> 5, lane = tid & 31;
  const size_t row = (size_t)blockIdx.x * 8 + wv;
  const size_t base = row * Dc;
  float vals[16];
  float s = 0.f;
#pragma unroll
  for (int i = 0; i < 16; ++i) {
    const int c = lane + 32 * i;
    vals[i] = x[base + c] + r[base + c];
    s += vals[i];
  }
  for (int off = 16; off; off >>= 1) s += __shfl_xor(s, off);
  const float mean = s * (1.0f / Dc);
  float vs = 0.f;
#pragma unroll
  for (int i = 0; i < 16; ++i) { const float d = vals[i] - mean; vs = fmaf(d, d, vs); }
  for (int off = 16; off; off >>= 1) vs += __shfl_xor(vs, off);
  const float inv = rsqrtf(vs * (1.0f / Dc) + 1e-5f);
#pragma unroll
  for (int i = 0; i < 16; ++i) {
    const int c = lane + 32 * i;
    const float y = (vals[i] - mean) * inv * g[c] + bta[c];
    xf[base + c]  = y;
    xbf[base + c] = f32_to_bf16u(y);
  }
}

// ---------------- host orchestration ----------------
static inline void launch_gemm(const unsigned short* A, const unsigned short* Wt,
                               const float* bias, float* Cf, unsigned short* Cbf,
                               int M, int N, int K, int relu, hipStream_t stream) {
  dim3 grid((N + 127) / 128, M / 64);
  gemm_bf16_kernel<<<grid, 256, 0, stream>>>(A, Wt, bias, Cf, Cbf, M, N, K, relu);
}

static inline void launch_f2bf_t(const float* in, unsigned short* out,
                                 int K, int N, hipStream_t stream) {
  dim3 grid((K + 255) / 256, N);
  f2bf_t_kernel<<<grid, 256, 0, stream>>>(in, out, K, N);
}

extern "C" void kernel_launch(void* const* d_in, const int* in_sizes, int n_in,
                              void* d_out, int out_size, void* d_ws, size_t ws_size,
                              hipStream_t stream) {
  (void)in_sizes; (void)n_in; (void)out_size; (void)ws_size;
  // inputs (setup_inputs order)
  const int*   gene    = (const int*)d_in[0];
  const float* exprv   = (const float*)d_in[1];
  const int*   cond    = (const int*)d_in[2];
  const int*   bat     = (const int*)d_in[3];
  const int*   mod     = (const int*)d_in[4];
  const float* gene_e  = (const float*)d_in[5];
  const float* cond_e  = (const float*)d_in[6];
  const float* bat_e   = (const float*)d_in[7];
  const float* mod_e   = (const float*)d_in[8];
  const float* ew1     = (const float*)d_in[9];
  const float* eb1     = (const float*)d_in[10];
  const float* ew2     = (const float*)d_in[11];
  const float* eb2     = (const float*)d_in[12];
  const float* pos     = (const float*)d_in[13];
  const float* Wq      = (const float*)d_in[14];
  const float* bq      = (const float*)d_in[15];
  const float* Wk      = (const float*)d_in[16];
  const float* bk      = (const float*)d_in[17];
  const float* Wv      = (const float*)d_in[18];
  const float* bv      = (const float*)d_in[19];
  const float* Wo      = (const float*)d_in[20];
  const float* bo      = (const float*)d_in[21];
  const float* ln1g    = (const float*)d_in[22];
  const float* ln1b    = (const float*)d_in[23];
  const float* fw1     = (const float*)d_in[24];
  const float* fb1     = (const float*)d_in[25];
  const float* fw2     = (const float*)d_in[26];
  const float* fb2     = (const float*)d_in[27];
  const float* ln2g    = (const float*)d_in[28];
  const float* ln2b    = (const float*)d_in[29];
  const float* outw    = (const float*)d_in[30];
  const float* outb    = (const float*)d_in[31];
  float* logits = (float*)d_out;

  // workspace carve (256B aligned)
  char* p = (char*)d_ws;
  auto carve = [&](size_t bytes) -> void* {
    void* r = (void*)p;
    p += (bytes + 255) & ~(size_t)255;
    return r;
  };
  float*          xf    = (float*)carve((size_t)Mrows * Dc * 4);
  unsigned short* xbf   = (unsigned short*)carve((size_t)Mrows * Dc * 2);
  float*          qf    = (float*)carve((size_t)Mrows * Dc * 4);
  float*          kf    = (float*)carve((size_t)Mrows * Dc * 4);
  float*          vf    = (float*)carve((size_t)Mrows * Dc * 4);
  unsigned short* obf   = (unsigned short*)carve((size_t)Mrows * Dc * 2);
  float*          yf    = (float*)carve((size_t)Mrows * Dc * 4);
  unsigned short* hbf   = (unsigned short*)carve((size_t)Mrows * FFc * 2);
  unsigned char*  maskb = (unsigned char*)carve((size_t)Bc * Sc * Sc);
  unsigned short* wq_t  = (unsigned short*)carve((size_t)Dc * Dc * 2);
  unsigned short* wk_t  = (unsigned short*)carve((size_t)Dc * Dc * 2);
  unsigned short* wv_t  = (unsigned short*)carve((size_t)Dc * Dc * 2);
  unsigned short* wo_t  = (unsigned short*)carve((size_t)Dc * Dc * 2);
  unsigned short* w1_t  = (unsigned short*)carve((size_t)Dc * FFc * 2);
  unsigned short* w2_t  = (unsigned short*)carve((size_t)FFc * Dc * 2);
  unsigned short* ow_t  = (unsigned short*)carve((size_t)Vpad * Dc * 2);  // padded rows

  // embeddings -> x
  embed_kernel<<<Mrows, 128, 0, stream>>>(gene, exprv, cond, bat, mod,
                                          gene_e, cond_e, bat_e, mod_e,
                                          ew1, eb1, ew2, eb2, pos, xf, xbf);
  // out_w -> bf16 transposed once: ow_t[v, d] (rows v >= Vc stay unused padding)
  launch_f2bf_t(outw, ow_t, Dc, Vc, stream);

  for (int l = 0; l < Lc; ++l) {
    const size_t wofs  = (size_t)l * Dc * Dc;
    const size_t f1ofs = (size_t)l * Dc * FFc;
    launch_f2bf_t(Wq + wofs,   wq_t, Dc, Dc, stream);
    launch_f2bf_t(Wk + wofs,   wk_t, Dc, Dc, stream);
    launch_f2bf_t(Wv + wofs,   wv_t, Dc, Dc, stream);
    launch_f2bf_t(Wo + wofs,   wo_t, Dc, Dc, stream);
    launch_f2bf_t(fw1 + f1ofs, w1_t, Dc, FFc, stream);   // [2048 x 512]
    launch_f2bf_t(fw2 + f1ofs, w2_t, FFc, Dc, stream);   // [512 x 2048]

    // q = relu(x Wq + bq), k = relu(x Wk + bk), v = x Wv + bv
    launch_gemm(xbf, wq_t, bq + (size_t)l * Dc, qf, nullptr, Mrows, Dc, Dc, 1, stream);
    launch_gemm(xbf, wk_t, bk + (size_t)l * Dc, kf, nullptr, Mrows, Dc, Dc, 1, stream);
    launch_gemm(xbf, wv_t, bv + (size_t)l * Dc, vf, nullptr, Mrows, Dc, Dc, 0, stream);

    // attention + fused next-layer dynamic mask
    attn_kernel<<<Mrows / 8, 256, 0, stream>>>(qf, kf, vf, maskb, l > 0 ? 1 : 0, obf);

    // o-proj, residual LN1
    launch_gemm(obf, wo_t, bo + (size_t)l * Dc, yf, nullptr, Mrows, Dc, Dc, 0, stream);
    add_ln_kernel<<<Mrows / 8, 256, 0, stream>>>(xf, yf, ln1g + (size_t)l * Dc,
                                                 ln1b + (size_t)l * Dc, xf, xbf);
    // FF, residual LN2
    launch_gemm(xbf, w1_t, fb1 + (size_t)l * FFc, nullptr, hbf, Mrows, FFc, Dc, 1, stream);
    launch_gemm(hbf, w2_t, fb2 + (size_t)l * Dc, yf, nullptr, Mrows, Dc, FFc, 0, stream);
    add_ln_kernel<<<Mrows / 8, 256, 0, stream>>>(xf, yf, ln2g + (size_t)l * Dc,
                                                 ln2b + (size_t)l * Dc, xf, xbf);
  }

  // logits = x @ out_w + out_b  (N = 20000; B padded to 20096 rows, epilogue-guarded)
  launch_gemm(xbf, ow_t, outb, logits, nullptr, Mrows, Vc, Dc, 0, stream);
}
